// BiLSTM_CRF_75651553952386
// MI455X (gfx1250) — matrix operand order repaired and checked by reading, not compiled, not run on hardware
//
#include <hip/hip_runtime.h>
#include <math.h>

// ---- problem constants ----
#define Vv     32000
#define Ee     256
#define Hh     512
#define Bb     64
#define Tt     512
#define Kk     34
#define H2c    256
#define G4     1024   // 4*H2
#define KP     48     // K padded to 3 WMMA tiles
#define NEGV   (-10000.0f)
#define S_START 32
#define S_STOP  33

typedef __attribute__((ext_vector_type(16))) __bf16          v16bf;
typedef __attribute__((ext_vector_type(8)))  float           v8f;
typedef __attribute__((ext_vector_type(8)))  unsigned short  v8u16;

// ---- bf16 helpers (bit-level, no __bf16 scalar arithmetic) ----
__device__ __forceinline__ unsigned short f2bf(float f) {
  union { float f; unsigned u; } x; x.f = f;
  unsigned u = x.u + 0x7FFFu + ((x.u >> 16) & 1u);
  return (unsigned short)(u >> 16);
}
__device__ __forceinline__ float bf2f(unsigned short s) {
  union { unsigned u; float f; } x; x.u = ((unsigned)s) << 16;
  return x.f;
}
__device__ __forceinline__ v16bf pack16(v8u16 lo, v8u16 hi) {
  union { v16bf b; unsigned short u[16]; } r;
#pragma unroll
  for (int i = 0; i < 8; ++i) { r.u[i] = lo[i]; r.u[i + 8] = hi[i]; }
  return r.b;
}

// A fragment 16x32 bf16 from row-major [*, lda], rows m0..m0+15, k k0..k0+31.
// Lane l<16 holds row m0+l, K = k0+{0..7,16..23}; lane l+16 holds K = k0+{8..15,24..31}.
__device__ __forceinline__ v16bf load_a(const unsigned short* base, int lda,
                                        int m0, int k0, int lane) {
  const int half = lane >> 4, m = lane & 15;
  const unsigned short* p = base + (size_t)(m0 + m) * lda + k0 + half * 8;
  v8u16 lo = *(const v8u16*)p;
  v8u16 hi = *(const v8u16*)(p + 16);
  return pack16(lo, hi);
}
// B fragment 32x16 bf16: column n = n0+(lane&15) is row n of w (row-major over k, ld=ldb).
// lanes 0-15 hold K=k0+{0..7,16..23} ; lanes 16-31 hold K=k0+{8..15,24..31}.
__device__ __forceinline__ v16bf load_b(const unsigned short* w, int ldb,
                                        int n0, int k0, int lane) {
  const int half = lane >> 4, n = lane & 15;
  const unsigned short* p = w + (size_t)(n0 + n) * ldb + k0 + half * 8;
  v8u16 lo = *(const v8u16*)p;
  v8u16 hi = *(const v8u16*)(p + 16);
  return pack16(lo, hi);
}
__device__ __forceinline__ v8f wmma_bf16(v16bf a, v16bf b, v8f c) {
  return __builtin_amdgcn_wmma_f32_16x16x32_bf16(false, a, false, b, (short)0, c,
                                                 false, false);
}
__device__ __forceinline__ float sigm(float x) { return 1.0f / (1.0f + __expf(-x)); }

// ---------------- kernels ----------------

__global__ void k_zero(float* out) { out[0] = 0.0f; }

// x_bf16[t*B+b][e] = embed[sentence[b][t]][e]
__global__ __launch_bounds__(256) void k_embed(const int* __restrict__ sent,
                                               const float* __restrict__ table,
                                               unsigned short* __restrict__ xb) {
  const int row = blockIdx.x;          // row = t*64 + b
  const int t = row >> 6, b = row & 63;
  const int tok = sent[b * Tt + t];
  xb[(size_t)row * Ee + threadIdx.x] = f2bf(table[(size_t)tok * Ee + threadIdx.x]);
}

__global__ __launch_bounds__(256) void k_cvt(const float* __restrict__ s,
                                             unsigned short* __restrict__ d, int n) {
  const int i = blockIdx.x * 256 + threadIdx.x;
  if (i < n) d[i] = f2bf(s[i]);
}

// pad w_tag [34,512] -> bf16 [48,512], zero rows >=34
__global__ __launch_bounds__(256) void k_wtag(const float* __restrict__ s,
                                              unsigned short* __restrict__ d) {
  const int i = blockIdx.x * 256 + threadIdx.x;  // n = 48*512 = 24576
  const int r = i >> 9, c = i & 511;
  d[i] = (r < Kk) ? f2bf(s[r * Hh + c]) : (unsigned short)0;
}

// xw[d] = x @ w_ih[d]^T + b[d], M=32768 N=1024 K=256, bf16 out, f32 accum.
// block = 256 thr (8 waves, 2M x 4N), block tile 128x128, grid (256, 8, 2)
__global__ __launch_bounds__(256) void k_gemm_xw(
    const unsigned short* __restrict__ xb,
    const unsigned short* __restrict__ wf, const float* __restrict__ biasf,
    const unsigned short* __restrict__ wb, const float* __restrict__ biasb,
    unsigned short* __restrict__ xwf, unsigned short* __restrict__ xwb) {
  const int lane = threadIdx.x & 31, wave = threadIdx.x >> 5;
  const int wm = wave & 1, wn = wave >> 1;
  const unsigned short* w = blockIdx.z ? wb : wf;
  const float* bias = blockIdx.z ? biasb : biasf;
  unsigned short* out = blockIdx.z ? xwb : xwf;
  const int m0 = blockIdx.x * 128 + wm * 64;
  const int n0 = blockIdx.y * 128 + wn * 32;

  v8f acc[4][2];
#pragma unroll
  for (int i = 0; i < 4; ++i) { acc[i][0] = (v8f)0.0f; acc[i][1] = (v8f)0.0f; }

#pragma unroll
  for (int kk = 0; kk < 8; ++kk) {
    const int k0 = kk * 32;
    v16bf b0 = load_b(w, Ee, n0, k0, lane);
    v16bf b1 = load_b(w, Ee, n0 + 16, k0, lane);
#pragma unroll
    for (int i = 0; i < 4; ++i) {
      v16bf a = load_a(xb, Ee, m0 + i * 16, k0, lane);
      acc[i][0] = wmma_bf16(a, b0, acc[i][0]);
      acc[i][1] = wmma_bf16(a, b1, acc[i][1]);
    }
  }
  const int half = lane >> 4, nl = lane & 15;
#pragma unroll
  for (int j = 0; j < 2; ++j) {
    const int col = n0 + j * 16 + nl;
    const float bv = bias[col];
#pragma unroll
    for (int i = 0; i < 4; ++i) {
      const int r0 = m0 + i * 16 + 8 * half;
#pragma unroll
      for (int r = 0; r < 8; ++r)
        out[(size_t)(r0 + r) * G4 + col] = f2bf(acc[i][j][r] + bv);
    }
  }
}

// Recurrent LSTM. grid = 8 blocks (dir*4 + batch-group), 512 thr (16 waves).
// Each block owns 16 batch rows: h,c state in LDS; per step 16x1024x256 WMMA matmul.
__global__ __launch_bounds__(512) void k_lstm(
    const unsigned short* __restrict__ xwf, const unsigned short* __restrict__ xwb,
    const unsigned short* __restrict__ whhf, const unsigned short* __restrict__ whhb,
    const float* __restrict__ h0, const float* __restrict__ c0,
    unsigned short* __restrict__ lstm_out) {
  const int dir = blockIdx.x >> 2;
  const int b0 = (blockIdx.x & 3) * 16;
  const unsigned short* xw = dir ? xwb : xwf;
  const unsigned short* whh = dir ? whhb : whhf;

  __shared__ unsigned short h_lds[16 * 264];   // bf16, padded stride (16B aligned)
  __shared__ float          c_lds[16 * H2c];
  __shared__ unsigned short gates[16 * G4];    // bf16 gate matmul result

  const int tid = threadIdx.x;
  const int lane = tid & 31, wave = tid >> 5;

  for (int i = tid; i < 16 * H2c; i += 512) {
    const int m = i >> 8, j = i & 255;
    h_lds[m * 264 + j] = f2bf(h0[(size_t)dir * Bb * H2c + (b0 + m) * H2c + j]);
    c_lds[m * H2c + j] = c0[(size_t)dir * Bb * H2c + (b0 + m) * H2c + j];
  }
  __syncthreads();

  for (int s = 0; s < Tt; ++s) {
    const int t = dir ? (Tt - 1 - s) : s;

    // --- phase A: gates_pre = h @ w_hh^T  (M=16, N=1024, K=256) ---
    v8f acc[4];
#pragma unroll
    for (int i = 0; i < 4; ++i) acc[i] = (v8f)0.0f;
#pragma unroll
    for (int kk = 0; kk < 8; ++kk) {
      const int k0 = kk * 32;
      v16bf a = load_a(h_lds, 264, 0, k0, lane);
#pragma unroll
      for (int nt = 0; nt < 4; ++nt) {
        const int n0 = wave * 64 + nt * 16;
        v16bf bfr = load_b(whh, H2c, n0, k0, lane);
        acc[nt] = wmma_bf16(a, bfr, acc[nt]);
      }
    }
    const int half = lane >> 4, nl = lane & 15;
#pragma unroll
    for (int nt = 0; nt < 4; ++nt) {
      const int col = wave * 64 + nt * 16 + nl;
#pragma unroll
      for (int r = 0; r < 8; ++r)
        gates[(r + 8 * half) * G4 + col] = f2bf(acc[nt][r]);
    }
    __syncthreads();

    // --- phase B: fused gate nonlinearity + state update ---
    for (int i = tid; i < 16 * H2c; i += 512) {
      const int m = i >> 8, j = i & 255;
      const unsigned short* xr = xw + (size_t)(t * Bb + b0 + m) * G4;
      const float gi = bf2f(gates[m * G4 + j])        + bf2f(xr[j]);
      const float gf = bf2f(gates[m * G4 + 256 + j])  + bf2f(xr[256 + j]);
      const float gg = bf2f(gates[m * G4 + 512 + j])  + bf2f(xr[512 + j]);
      const float go = bf2f(gates[m * G4 + 768 + j])  + bf2f(xr[768 + j]);
      float c = sigm(gf) * c_lds[m * H2c + j] + sigm(gi) * tanhf(gg);
      float h = sigm(go) * tanhf(c);
      c_lds[m * H2c + j] = c;
      h_lds[m * 264 + j] = f2bf(h);
      lstm_out[(size_t)(t * Bb + b0 + m) * Hh + dir * H2c + j] = f2bf(h);
    }
    // prefetch next step's xw slice (16 rows x 1024 bf16 = 256 cachelines)
    if (s + 1 < Tt && tid < 256) {
      const int tn = dir ? (Tt - 2 - s) : (s + 1);
      __builtin_prefetch(xw + (size_t)(tn * Bb + b0) * G4 + (size_t)tid * 64, 0, 3);
    }
    __syncthreads();
  }
}

// feats = lstm_out @ w_tag^T + b_tag. M=32768 K=512 N=48(pad). grid (64,3), 256 thr.
__global__ __launch_bounds__(256) void k_feats(const unsigned short* __restrict__ lstm,
                                               const unsigned short* __restrict__ wtag,
                                               const float* __restrict__ btag,
                                               float* __restrict__ feats) {
  const int lane = threadIdx.x & 31, wave = threadIdx.x >> 5;
  const int m0 = blockIdx.x * 512 + wave * 64;
  const int n0 = blockIdx.y * 16;
  v8f acc[4];
#pragma unroll
  for (int i = 0; i < 4; ++i) acc[i] = (v8f)0.0f;
#pragma unroll
  for (int kk = 0; kk < 16; ++kk) {
    const int k0 = kk * 32;
    v16bf b = load_b(wtag, Hh, n0, k0, lane);
#pragma unroll
    for (int i = 0; i < 4; ++i) {
      v16bf a = load_a(lstm, Hh, m0 + i * 16, k0, lane);
      acc[i] = wmma_bf16(a, b, acc[i]);
    }
  }
  const int half = lane >> 4, col = n0 + (lane & 15);
  const float bv = (col < Kk) ? btag[col] : 0.0f;
#pragma unroll
  for (int i = 0; i < 4; ++i) {
    const int r0 = m0 + i * 16 + 8 * half;
#pragma unroll
    for (int r = 0; r < 8; ++r)
      feats[(size_t)(r0 + r) * KP + col] = acc[i][r] + bv;
  }
}

// CRF forward scan + gold path score. One wave32 per batch element.
// grid 2 x 1024 thr (32 waves/block).
__global__ __launch_bounds__(1024) void k_crf(const float* __restrict__ feats,
                                              const int* __restrict__ tags,
                                              const float* __restrict__ transitions,
                                              float* __restrict__ out) {
  __shared__ float tr[Kk * Kk];
  __shared__ float alpha[32][36];
  const int tid = threadIdx.x;
  for (int i = tid; i < Kk * Kk; i += 1024) {
    const int n = i / Kk, p = i % Kk;
    tr[i] = (n == S_START || p == S_STOP) ? NEGV : transitions[i];
  }
  const int wave = tid >> 5, lane = tid & 31;
  const int b = blockIdx.x * 32 + wave;
  alpha[wave][lane] = NEGV;                           // states 0..31
  if (lane < 2) alpha[wave][32 + lane] = (lane == 0) ? 0.0f : NEGV;  // START=32 -> 0
  __syncthreads();

  float* al = alpha[wave];
  for (int t = 0; t < Tt; ++t) {
    const float* fr = feats + (size_t)(t * Bb + b) * KP;
    // state n1 = lane (0..31)
    float m1 = -1e30f;
    for (int p = 0; p < Kk; ++p) m1 = fmaxf(m1, al[p] + tr[lane * Kk + p]);
    float s1 = 0.0f;
    for (int p = 0; p < Kk; ++p) s1 += __expf(al[p] + tr[lane * Kk + p] - m1);
    const float new1 = m1 + __logf(s1) + fr[lane];
    // states 32,33 on lanes 0,1
    float new2 = 0.0f;
    if (lane < 2) {
      const int n2 = 32 + lane;
      float m2 = -1e30f;
      for (int p = 0; p < Kk; ++p) m2 = fmaxf(m2, al[p] + tr[n2 * Kk + p]);
      float s2 = 0.0f;
      for (int p = 0; p < Kk; ++p) s2 += __expf(al[p] + tr[n2 * Kk + p] - m2);
      new2 = m2 + __logf(s2) + fr[n2];
    }
    al[lane] = new1;                 // all reads of old alpha happen above (same wave)
    if (lane < 2) al[32 + lane] = new2;
  }
  // log_z = logsumexp(alpha + trans[STOP,:])
  float m = -1e30f;
  for (int p = 0; p < Kk; ++p) m = fmaxf(m, al[p] + tr[S_STOP * Kk + p]);
  float s = 0.0f;
  for (int p = 0; p < Kk; ++p) s += __expf(al[p] + tr[S_STOP * Kk + p] - m);
  const float logz = m + __logf(s);
  // gold score, t distributed across lanes
  float g = 0.0f;
  for (int t = lane; t < Tt; t += 32) {
    const int cur = tags[b * Tt + t];
    const int prev = (t == 0) ? S_START : tags[b * Tt + t - 1];
    g += tr[cur * Kk + prev] + feats[(size_t)(t * Bb + b) * KP + cur];
  }
  for (int off = 16; off > 0; off >>= 1) g += __shfl_down(g, off, 32);
  if (lane == 0) {
    g += tr[S_STOP * Kk + tags[b * Tt + Tt - 1]];
    atomicAdd(out, logz - g);
  }
}

// ---------------- host launch ----------------
extern "C" void kernel_launch(void* const* d_in, const int* in_sizes, int n_in,
                              void* d_out, int out_size, void* d_ws, size_t ws_size,
                              hipStream_t stream) {
  const int*   sentence    = (const int*)d_in[0];
  const int*   tags        = (const int*)d_in[1];
  const float* embed       = (const float*)d_in[2];
  const float* w_ih_f      = (const float*)d_in[3];
  const float* w_hh_f      = (const float*)d_in[4];
  const float* b_f         = (const float*)d_in[5];
  const float* w_ih_b      = (const float*)d_in[6];
  const float* w_hh_b      = (const float*)d_in[7];
  const float* b_b         = (const float*)d_in[8];
  const float* h0          = (const float*)d_in[9];
  const float* c0          = (const float*)d_in[10];
  const float* w_tag       = (const float*)d_in[11];
  const float* b_tag       = (const float*)d_in[12];
  const float* transitions = (const float*)d_in[13];
  float* out = (float*)d_out;

  char* ws = (char*)d_ws;
  size_t off = 0;
  auto alloc = [&](size_t bytes) -> void* {
    void* p = ws + off;
    off += (bytes + 255) & ~(size_t)255;
    return p;
  };
  const size_t TB = (size_t)Tt * Bb;  // 32768
  unsigned short* x_bf     = (unsigned short*)alloc(TB * Ee * 2);
  unsigned short* xw_f     = (unsigned short*)alloc(TB * G4 * 2);
  unsigned short* xw_b     = (unsigned short*)alloc(TB * G4 * 2);
  unsigned short* lstm_out = (unsigned short*)alloc(TB * Hh * 2);
  float*          feats    = (float*)alloc(TB * KP * 4);
  unsigned short* wihf_bf  = (unsigned short*)alloc((size_t)G4 * Ee * 2);
  unsigned short* wihb_bf  = (unsigned short*)alloc((size_t)G4 * Ee * 2);
  unsigned short* whhf_bf  = (unsigned short*)alloc((size_t)G4 * H2c * 2);
  unsigned short* whhb_bf  = (unsigned short*)alloc((size_t)G4 * H2c * 2);
  unsigned short* wtag_bf  = (unsigned short*)alloc((size_t)KP * Hh * 2);

  k_zero<<<1, 1, 0, stream>>>(out);
  k_embed<<<(int)TB, 256, 0, stream>>>(sentence, embed, x_bf);

  const int nW = G4 * Ee;  // 262144
  k_cvt<<<nW / 256, 256, 0, stream>>>(w_ih_f, wihf_bf, nW);
  k_cvt<<<nW / 256, 256, 0, stream>>>(w_ih_b, wihb_bf, nW);
  k_cvt<<<nW / 256, 256, 0, stream>>>(w_hh_f, whhf_bf, nW);
  k_cvt<<<nW / 256, 256, 0, stream>>>(w_hh_b, whhb_bf, nW);
  k_wtag<<<(KP * Hh) / 256, 256, 0, stream>>>(w_tag, wtag_bf);

  k_gemm_xw<<<dim3(256, 8, 2), 256, 0, stream>>>(x_bf, wihf_bf, b_f, wihb_bf, b_b,
                                                 xw_f, xw_b);
  k_lstm<<<8, 512, 0, stream>>>(xw_f, xw_b, whhf_bf, whhb_bf, h0, c0, lstm_out);
  k_feats<<<dim3(64, 3), 256, 0, stream>>>(lstm_out, wtag_bf, b_tag, feats);
  k_crf<<<2, 1024, 0, stream>>>(feats, tags, transitions, out);
}
